// TransformerEncoderLayer_58213986730308
// MI455X (gfx1250) — matrix operand order repaired and checked by reading, not compiled, not run on hardware
//
#include <hip/hip_runtime.h>

// ---------------------------------------------------------------------------
// Problem constants (match reference)
// ---------------------------------------------------------------------------
#define S_DIM  1024
#define B_DIM  8
#define D_DIM  1024
#define H_DIM  16
#define DK_DIM 64
#define F_DIM  4096
#define T_DIM  (S_DIM * B_DIM)   // 8192 tokens
#define LN_EPS 1e-5f

typedef __bf16 bf16;
typedef __attribute__((ext_vector_type(16))) __bf16 v16bf;
typedef __attribute__((ext_vector_type(8)))  float  v8f;
typedef __attribute__((ext_vector_type(4)))  unsigned uV4;   // AS1-safe 16B vector
typedef __attribute__((ext_vector_type(4)))  float    fV4;

__device__ __forceinline__ unsigned short f2bf_bits(float f) {
  unsigned u = __builtin_bit_cast(unsigned, f);
  return (unsigned short)((u + 0x7FFFu + ((u >> 16) & 1u)) >> 16);  // RNE
}
__device__ __forceinline__ bf16 f2bf(float f) {
  return __builtin_bit_cast(bf16, f2bf_bits(f));
}

union FragU  { uV4 u4[2]; v16bf v; };
union Pack8  { unsigned short u[8]; uV4 v; };

// ---- explicit global-address-space accessors (avoid flat_* lowering) -------
#define GLD_U4(p) (*(const __attribute__((address_space(1))) uV4*)(unsigned long long)(p))
#define GLD_F4(p) (*(const __attribute__((address_space(1))) fV4*)(unsigned long long)(p))
#define GST_F(p, x)  (*(__attribute__((address_space(1))) float*)(unsigned long long)(p) = (x))
#define GST_BF(p, x) (*(__attribute__((address_space(1))) bf16*)(unsigned long long)(p)  = (x))

// ---- CDNA5 async global->LDS copy (ASYNCcnt-tracked), with fallback --------
#if defined(__has_builtin)
# if __has_builtin(__builtin_amdgcn_global_load_async_to_lds_b128) && \
     __has_builtin(__builtin_amdgcn_s_wait_asynccnt)
#  define HAVE_ASYNC_LDS 1
# endif
#endif
#ifndef HAVE_ASYNC_LDS
# define HAVE_ASYNC_LDS 0
#endif

#if HAVE_ASYNC_LDS
// Builtin params (from clang diagnostic): non-const pointers to
// `int __attribute__((vector_size(16)))`; arg0 = global src, arg1 = LDS dst.
typedef int v4i_vs __attribute__((vector_size(16)));
// flat LDS address low 32 bits == LDS byte offset (ISA 10.2 aperture table)
#define TO_GBL(p)  ((__attribute__((address_space(1))) v4i_vs*)(unsigned long long)(p))
#define TO_LDS(p)  ((__attribute__((address_space(3))) v4i_vs*)(unsigned)(unsigned long long)(p))
#endif

// ---------------------------------------------------------------------------
// Generic bf16 WMMA GEMM:  C[z] = epilogue(alpha * A[z] @ B[z] + bias)
//   128x128 block tile, K-step 32, 256 threads = 8 waves (4M x 2N), each wave
//   a 32x64 tile = 2x4 v_wmma_f32_16x16x32_bf16 accumulators.
//   LDS ping-pong double buffer: stage tile s+1 while computing tile s
//   (one barrier per K-step; global latency hidden behind the 8 WMMAs).
//   No-transpose/no-convert staging uses GLOBAL_LOAD_ASYNC_TO_LDS_B128.
//   LDS row stride padded to 40 bf16 (80B): 16 fragment lanes hit 16 distinct
//   banks (20*m mod 64 is a permutation) and reads stay 16B-aligned.
//   Fragment feeds (ISA 7.12.2), two ds_load_b128 per fragment:
//     A[m][k]: lane m=lane&15, K {kb..kb+7, 16+kb..}, kb=(lane>>4)*8
//     B[n][k]: lane n=lane&15, K {kb..kb+7, 16+kb..}, same kb as A
//   EPI: 0 none, 1 +bias, 2 relu(+bias); B_TRANS: B given [N,K] row-major;
//   A_F32: A is fp32 in global, converted to bf16 while staging.
// ---------------------------------------------------------------------------
#define BM 128
#define BN 128
#define BK 32
#define LSTR 40   // padded LDS row stride in bf16 elements

template<int EPI, bool OUT_BF16, bool B_TRANS, bool A_F32>
__global__ __launch_bounds__(256)
void gemm_wmma(const void* __restrict__ Aptr, long lda, long sAb, long sAh,
               const void* __restrict__ Bptr, long ldb, long sBb, long sBh,
               const float* __restrict__ bias,
               void* __restrict__ Cptr, long ldc, long sCb, long sCh,
               int M, int N, int Kdim, int Hh, float alpha)
{
  __shared__ bf16 As[2][BM * LSTR];   // [m][k], padded rows
  __shared__ bf16 Bs[2][BN * LSTR];   // [n][k], padded rows

  const int tid  = threadIdx.x;
  const int lane = tid & 31;
  const int wave = tid >> 5;
  const int wm   = wave & 3;     // 4 waves along M
  const int wn   = wave >> 2;    // 2 waves along N

  const int z  = blockIdx.z;
  const int bI = z / Hh;
  const int hI = z % Hh;
  const long offA = (long)bI * sAb + (long)hI * sAh;
  const long offB = (long)bI * sBb + (long)hI * sBh;
  const long offC = (long)bI * sCb + (long)hI * sCh;

  const int m0 = blockIdx.y * BM;
  const int n0 = blockIdx.x * BN;

  const int l15  = lane & 15;
  const int kb8  = (lane >> 4) * 8;    // A-fragment K base
  const int kh16 = (lane >> 4) * 16;   // unused; B fragment uses kb8

  // staging registers for fallback / conversion paths (2 chunks per thread)
  uV4 aStg[2];
  fV4 aStgF[2][2];
  uV4 bStg[2];
  const uV4 uzero = {0u, 0u, 0u, 0u};

  // ---- stage issue: start global reads for K-tile k0 into LDS buffer `buf`
  auto issueA = [&](int k0, int buf) {
    if constexpr (A_F32) {
      const float* Ag = (const float*)Aptr + offA;
#pragma unroll
      for (int t = 0; t < 2; ++t) {
        const int c = tid + t * 256, row = c >> 2, col = (c & 3) * 8;
        const float* s = Ag + (long)(m0 + row) * lda + (k0 + col);
        aStgF[t][0] = GLD_F4(s);
        aStgF[t][1] = GLD_F4(s + 4);
      }
    } else {
      const bf16* Ag = (const bf16*)Aptr + offA;
#pragma unroll
      for (int t = 0; t < 2; ++t) {
        const int c = tid + t * 256, row = c >> 2, col = (c & 3) * 8;
        const bf16* s = Ag + (long)(m0 + row) * lda + (k0 + col);
#if HAVE_ASYNC_LDS
        __builtin_amdgcn_global_load_async_to_lds_b128(
            TO_GBL(s), TO_LDS(&As[buf][row * LSTR + col]), 0, 0);
#else
        aStg[t] = GLD_U4(s);
#endif
      }
    }
    (void)buf;
  };
  auto issueB = [&](int k0, int buf) {
    const bf16* Bg = (const bf16*)Bptr + offB;
    if constexpr (B_TRANS) {          // B given [N,K] row-major
#pragma unroll
      for (int t = 0; t < 2; ++t) {
        const int c = tid + t * 256, n = c >> 2, col = (c & 3) * 8;
#if HAVE_ASYNC_LDS
        if (n0 + n < N)
          __builtin_amdgcn_global_load_async_to_lds_b128(
              TO_GBL(Bg + (long)(n0 + n) * ldb + (k0 + col)),
              TO_LDS(&Bs[buf][n * LSTR + col]), 0, 0);
        else
          *(uV4*)&Bs[buf][n * LSTR + col] = uzero;
#else
        uV4 v = uzero;
        if (n0 + n < N) v = GLD_U4(Bg + (long)(n0 + n) * ldb + (k0 + col));
        bStg[t] = v;
#endif
      }
    } else {                          // B given [K,N] row-major: one b128/chunk
#pragma unroll
      for (int t = 0; t < 2; ++t) {
        const int c = tid + t * 256, kk = c >> 4, nc = (c & 15) * 8;
        uV4 v = uzero;
        if (n0 + nc < N) v = GLD_U4(Bg + (long)(k0 + kk) * ldb + (n0 + nc));
        bStg[t] = v;
      }
    }
    (void)buf;
  };
  // ---- stage commit: move staged registers into LDS buffer `buf`
  auto commitA = [&](int buf) {
    if constexpr (A_F32) {
#pragma unroll
      for (int t = 0; t < 2; ++t) {
        const int c = tid + t * 256, row = c >> 2, col = (c & 3) * 8;
        Pack8 pk;
#pragma unroll
        for (int j = 0; j < 4; ++j) {
          pk.u[j]     = f2bf_bits(aStgF[t][0][j]);
          pk.u[4 + j] = f2bf_bits(aStgF[t][1][j]);
        }
        *(uV4*)&As[buf][row * LSTR + col] = pk.v;
      }
    } else {
#if !HAVE_ASYNC_LDS
#pragma unroll
      for (int t = 0; t < 2; ++t) {
        const int c = tid + t * 256, row = c >> 2, col = (c & 3) * 8;
        *(uV4*)&As[buf][row * LSTR + col] = aStg[t];
      }
#endif
    }
    (void)buf;
  };
  auto commitB = [&](int buf) {
    if constexpr (B_TRANS) {
#if !HAVE_ASYNC_LDS
#pragma unroll
      for (int t = 0; t < 2; ++t) {
        const int c = tid + t * 256, n = c >> 2, col = (c & 3) * 8;
        *(uV4*)&Bs[buf][n * LSTR + col] = bStg[t];
      }
#endif
    } else {                          // scatter-transpose into [n][k]
#pragma unroll
      for (int t = 0; t < 2; ++t) {
        const int c = tid + t * 256, kk = c >> 4, nc = (c & 15) * 8;
        Pack8 pk; pk.v = bStg[t];
#pragma unroll
        for (int j = 0; j < 8; ++j)
          Bs[buf][(nc + j) * LSTR + kk] = __builtin_bit_cast(bf16, pk.u[j]);
      }
    }
    (void)buf;
  };

  v8f acc[2][4];
  const v8f vzero = {0.f, 0.f, 0.f, 0.f, 0.f, 0.f, 0.f, 0.f};
#pragma unroll
  for (int i = 0; i < 2; ++i)
#pragma unroll
    for (int j = 0; j < 4; ++j) acc[i][j] = vzero;

  // ---- prologue: stage tile 0 into buffer 0 ----
  issueA(0, 0); issueB(0, 0);
  commitA(0);   commitB(0);
#if HAVE_ASYNC_LDS
  __builtin_amdgcn_s_wait_asynccnt(0);
#endif
  __syncthreads();

  const int nsteps = Kdim / BK;
  for (int s = 0; s < nsteps; ++s) {
    const int cur = s & 1, nxt = cur ^ 1;
    const bool more = (s + 1) < nsteps;
    if (more) { issueA((s + 1) * BK, nxt); issueB((s + 1) * BK, nxt); }

    // ---- fragments from buffer `cur` + 8 WMMAs ----
    FragU af[2], bfr[4];
#pragma unroll
    for (int i = 0; i < 2; ++i) {
      const int m = wm * 32 + i * 16 + l15;
      af[i].u4[0] = *(const uV4*)&As[cur][m * LSTR + kb8];
      af[i].u4[1] = *(const uV4*)&As[cur][m * LSTR + 16 + kb8];
    }
#pragma unroll
    for (int j = 0; j < 4; ++j) {
      const int n = wn * 64 + j * 16 + l15;
      bfr[j].u4[0] = *(const uV4*)&Bs[cur][n * LSTR + kb8];
      bfr[j].u4[1] = *(const uV4*)&Bs[cur][n * LSTR + 16 + kb8];
    }
#pragma unroll
    for (int i = 0; i < 2; ++i)
#pragma unroll
      for (int j = 0; j < 4; ++j)
        acc[i][j] = __builtin_amdgcn_wmma_f32_16x16x32_bf16(
            false, af[i].v, false, bfr[j].v, (short)0, acc[i][j], false, false);

    if (more) { commitA(nxt); commitB(nxt); }
#if HAVE_ASYNC_LDS
    if (more) __builtin_amdgcn_s_wait_asynccnt(0);
#endif
    __syncthreads();
  }

  // ---- epilogue: C lane layout: row r -> m = r + (lane>>4)*8, n = lane&15 ----
  const int mhalf = (lane >> 4) * 8;
#pragma unroll
  for (int j = 0; j < 4; ++j) {
    const int n = n0 + wn * 64 + j * 16 + l15;
    if (n >= N) continue;
    float bv = 0.f;
    if constexpr (EPI >= 1) bv = bias[n];
#pragma unroll
    for (int i = 0; i < 2; ++i) {
#pragma unroll
      for (int r = 0; r < 8; ++r) {
        const int m = m0 + wm * 32 + i * 16 + mhalf + r;
        float v = acc[i][j][r] * alpha + bv;
        if constexpr (EPI == 2) v = fmaxf(v, 0.f);
        const long idx = offC + (long)m * ldc + n;
        if constexpr (OUT_BF16) GST_BF((bf16*)Cptr + idx, f2bf(v));
        else                    GST_F((float*)Cptr + idx, v);
      }
    }
  }
}

// ---------------------------------------------------------------------------
// Reductions (wave32 shuffles + LDS across the 8 waves of a 256-thread block)
// ---------------------------------------------------------------------------
__device__ __forceinline__ float waveSum(float v) {
#pragma unroll
  for (int o = 16; o > 0; o >>= 1) v += __shfl_xor(v, o);
  return v;
}
__device__ __forceinline__ float waveMax(float v) {
#pragma unroll
  for (int o = 16; o > 0; o >>= 1) v = fmaxf(v, __shfl_xor(v, o));
  return v;
}
__device__ __forceinline__ float blockSum(float v, float* sh) {
  v = waveSum(v);
  if ((threadIdx.x & 31) == 0) sh[threadIdx.x >> 5] = v;
  __syncthreads();
  float t = 0.f;
#pragma unroll
  for (int i = 0; i < 8; ++i) t += sh[i];
  __syncthreads();
  return t;
}
__device__ __forceinline__ float blockMax(float v, float* sh) {
  v = waveMax(v);
  if ((threadIdx.x & 31) == 0) sh[threadIdx.x >> 5] = v;
  __syncthreads();
  float t = sh[0];
#pragma unroll
  for (int i = 1; i < 8; ++i) t = fmaxf(t, sh[i]);
  __syncthreads();
  return t;
}

// ---------------------------------------------------------------------------
// Elementwise / row kernels (vectorized: 4 fp32 in, packed 4x bf16 out)
// ---------------------------------------------------------------------------
__global__ __launch_bounds__(256)
void prep_qk_kernel(const float4* __restrict__ src, const float4* __restrict__ pos,
                    ushort4* __restrict__ qk, long n4) {
  const long i = (long)blockIdx.x * 256 + threadIdx.x;
  if (i >= n4) return;
  const float4 a = src[i], b = pos[i];
  ushort4 o;
  o.x = f2bf_bits(a.x + b.x); o.y = f2bf_bits(a.y + b.y);
  o.z = f2bf_bits(a.z + b.z); o.w = f2bf_bits(a.w + b.w);
  qk[i] = o;
}

__global__ __launch_bounds__(256)
void f2bf_kernel(const float4* __restrict__ in, ushort4* __restrict__ out, long n4) {
  const long i = (long)blockIdx.x * 256 + threadIdx.x;
  if (i >= n4) return;
  const float4 a = in[i];
  ushort4 o;
  o.x = f2bf_bits(a.x); o.y = f2bf_bits(a.y);
  o.z = f2bf_bits(a.z); o.w = f2bf_bits(a.w);
  out[i] = o;
}

// one block per attention row of length S_DIM; fp32 softmax in place (d_out)
__global__ __launch_bounds__(256)
void softmax_kernel(float* __restrict__ attn) {
  __shared__ float sh[8];
  float* p = attn + (long)blockIdx.x * S_DIM;
  const int tid = threadIdx.x;
  float x[S_DIM / 256];
  float m = -3.402823466e38f;
#pragma unroll
  for (int i = 0; i < S_DIM / 256; ++i) {
    x[i] = p[tid + i * 256];
    m = fmaxf(m, x[i]);
  }
  m = blockMax(m, sh);
  float s = 0.f;
#pragma unroll
  for (int i = 0; i < S_DIM / 256; ++i) {
    x[i] = __expf(x[i] - m);
    s += x[i];
  }
  s = blockSum(s, sh);
  const float inv = 1.f / s;
#pragma unroll
  for (int i = 0; i < S_DIM / 256; ++i) p[tid + i * 256] = x[i] * inv;
}

// one block per token row of length D_DIM: out = LN(X + Y) * g + b
__global__ __launch_bounds__(256)
void residual_ln_kernel(const float* __restrict__ X, const float* __restrict__ Y,
                        const float* __restrict__ g, const float* __restrict__ b,
                        float* __restrict__ outF, bf16* __restrict__ outB) {
  __shared__ float sh[8];
  const long row = blockIdx.x;
  const int tid = threadIdx.x;
  const float* xp = X + row * D_DIM;
  const float* yp = Y + row * D_DIM;
  float v[D_DIM / 256];
  float s = 0.f;
#pragma unroll
  for (int i = 0; i < D_DIM / 256; ++i) {
    v[i] = xp[tid + i * 256] + yp[tid + i * 256];
    s += v[i];
  }
  s = blockSum(s, sh);
  const float mean = s * (1.f / D_DIM);
  float q = 0.f;
#pragma unroll
  for (int i = 0; i < D_DIM / 256; ++i) {
    const float d = v[i] - mean;
    q += d * d;
  }
  q = blockSum(q, sh);
  const float rstd = rsqrtf(q * (1.f / D_DIM) + LN_EPS);
#pragma unroll
  for (int i = 0; i < D_DIM / 256; ++i) {
    const int d = tid + i * 256;
    const float o = (v[i] - mean) * rstd * g[d] + b[d];
    outF[row * D_DIM + d] = o;
    if (outB) outB[row * D_DIM + d] = f2bf(o);
  }
}

// ---------------------------------------------------------------------------
// Launch: full encoder layer.  Token layout everywhere is [S,B,D] flattened to
// [T=8192, D]; per-(b,h) attention slices are strided views (lda = B*D,
// base = b*D + h*64) so no transposes are needed.  ~280 MiB workspace.
// ---------------------------------------------------------------------------
extern "C" void kernel_launch(void* const* d_in, const int* in_sizes, int n_in,
                              void* d_out, int out_size, void* d_ws, size_t ws_size,
                              hipStream_t stream) {
  const float* src = (const float*)d_in[0];
  const float* pos = (const float*)d_in[1];
  const float* Wq  = (const float*)d_in[2];
  const float* bq  = (const float*)d_in[3];
  const float* Wk  = (const float*)d_in[4];
  const float* bk  = (const float*)d_in[5];
  const float* Wv  = (const float*)d_in[6];
  const float* bv  = (const float*)d_in[7];
  const float* Wo  = (const float*)d_in[8];
  const float* bo  = (const float*)d_in[9];
  const float* W1  = (const float*)d_in[10];
  const float* b1  = (const float*)d_in[11];
  const float* W2  = (const float*)d_in[12];
  const float* b2  = (const float*)d_in[13];
  const float* g1  = (const float*)d_in[14];
  const float* be1 = (const float*)d_in[15];
  const float* g2  = (const float*)d_in[16];
  const float* be2 = (const float*)d_in[17];

  // ---- workspace carve-up ----
  char* w = (char*)d_ws;
  bf16* WqB = (bf16*)w; w += (size_t)D_DIM * D_DIM * 2;
  bf16* WkB = (bf16*)w; w += (size_t)D_DIM * D_DIM * 2;
  bf16* WvB = (bf16*)w; w += (size_t)D_DIM * D_DIM * 2;
  bf16* WoB = (bf16*)w; w += (size_t)D_DIM * D_DIM * 2;
  bf16* W1B = (bf16*)w; w += (size_t)D_DIM * F_DIM * 2;
  bf16* W2B = (bf16*)w; w += (size_t)F_DIM * D_DIM * 2;
  bf16* qkB = (bf16*)w; w += (size_t)T_DIM * D_DIM * 2;
  bf16* QB  = (bf16*)w; w += (size_t)T_DIM * D_DIM * 2;
  bf16* KB  = (bf16*)w; w += (size_t)T_DIM * D_DIM * 2;
  bf16* VB  = (bf16*)w; w += (size_t)T_DIM * D_DIM * 2;
  bf16* ctxB = (bf16*)w; w += (size_t)T_DIM * D_DIM * 2;
  float* xF  = (float*)w; w += (size_t)T_DIM * D_DIM * 4;
  bf16*  xB  = (bf16*)w;  w += (size_t)T_DIM * D_DIM * 2;
  float* aoF = (float*)w; w += (size_t)T_DIM * D_DIM * 4;
  float* ffF = (float*)w; w += (size_t)T_DIM * D_DIM * 4;
  bf16*  h1B = (bf16*)w;  w += (size_t)T_DIM * F_DIM * 2;

  float* outF  = (float*)d_out;                          // [S,B,D]
  float* attnF = (float*)d_out + (size_t)T_DIM * D_DIM;  // [B,H,S,S]

  const long nTok = (long)T_DIM * D_DIM;
  const dim3 blk(256);

  // ---- precision prep (vectorized x4) ----
  prep_qk_kernel<<<dim3((nTok / 4 + 255) / 256), blk, 0, stream>>>(
      (const float4*)src, (const float4*)pos, (ushort4*)qkB, nTok / 4);
  const long nDD = (long)D_DIM * D_DIM / 4, nDF = (long)D_DIM * F_DIM / 4;
  f2bf_kernel<<<dim3((nDD + 255) / 256), blk, 0, stream>>>((const float4*)Wq, (ushort4*)WqB, nDD);
  f2bf_kernel<<<dim3((nDD + 255) / 256), blk, 0, stream>>>((const float4*)Wk, (ushort4*)WkB, nDD);
  f2bf_kernel<<<dim3((nDD + 255) / 256), blk, 0, stream>>>((const float4*)Wv, (ushort4*)WvB, nDD);
  f2bf_kernel<<<dim3((nDD + 255) / 256), blk, 0, stream>>>((const float4*)Wo, (ushort4*)WoB, nDD);
  f2bf_kernel<<<dim3((nDF + 255) / 256), blk, 0, stream>>>((const float4*)W1, (ushort4*)W1B, nDF);
  f2bf_kernel<<<dim3((nDF + 255) / 256), blk, 0, stream>>>((const float4*)W2, (ushort4*)W2B, nDF);

  // ---- Q/K/V projections: [T,D] @ [D,D] + bias ----
  const dim3 gDD(D_DIM / BN, T_DIM / BM, 1);
  gemm_wmma<1, true, false, false><<<gDD, blk, 0, stream>>>(
      qkB, D_DIM, 0, 0, WqB, D_DIM, 0, 0, bq, QB, D_DIM, 0, 0,
      T_DIM, D_DIM, D_DIM, 1, 1.f);
  gemm_wmma<1, true, false, false><<<gDD, blk, 0, stream>>>(
      qkB, D_DIM, 0, 0, WkB, D_DIM, 0, 0, bk, KB, D_DIM, 0, 0,
      T_DIM, D_DIM, D_DIM, 1, 1.f);
  gemm_wmma<1, true, false, true><<<gDD, blk, 0, stream>>>(   // value input = raw src (fp32)
      src, D_DIM, 0, 0, WvB, D_DIM, 0, 0, bv, VB, D_DIM, 0, 0,
      T_DIM, D_DIM, D_DIM, 1, 1.f);

  // ---- scores = Q @ K^T / sqrt(DK), fp32 straight into d_out ----
  const long ldTok = (long)B_DIM * D_DIM;
  gemm_wmma<0, false, true, false><<<dim3(S_DIM / BN, S_DIM / BM, B_DIM * H_DIM), blk, 0, stream>>>(
      QB, ldTok, D_DIM, DK_DIM,
      KB, ldTok, D_DIM, DK_DIM, nullptr,
      attnF, S_DIM, (long)H_DIM * S_DIM * S_DIM, (long)S_DIM * S_DIM,
      S_DIM, S_DIM, DK_DIM, H_DIM, 0.125f);

  // ---- softmax in place over d_out attn region ----
  softmax_kernel<<<dim3(B_DIM * H_DIM * S_DIM), blk, 0, stream>>>(attnF);

  // ---- ctx = attn @ V  (A read fp32 from d_out, converted while staging) ----
  gemm_wmma<0, true, false, true><<<dim3(1, S_DIM / BM, B_DIM * H_DIM), blk, 0, stream>>>(
      attnF, S_DIM, (long)H_DIM * S_DIM * S_DIM, (long)S_DIM * S_DIM,
      VB, ldTok, D_DIM, DK_DIM, nullptr,
      ctxB, ldTok, D_DIM, DK_DIM,
      S_DIM, DK_DIM, S_DIM, H_DIM, 1.f);

  // ---- attn_out = ctx @ Wo + bo (fp32) ----
  gemm_wmma<1, false, false, false><<<gDD, blk, 0, stream>>>(
      ctxB, D_DIM, 0, 0, WoB, D_DIM, 0, 0, bo, aoF, D_DIM, 0, 0,
      T_DIM, D_DIM, D_DIM, 1, 1.f);

  // ---- x = LN(src + attn_out) ----
  residual_ln_kernel<<<dim3(T_DIM), blk, 0, stream>>>(src, aoF, g1, be1, xF, xB);

  // ---- FFN: relu(x @ W1 + b1) @ W2 + b2 ----
  gemm_wmma<2, true, false, false><<<dim3(F_DIM / BN, T_DIM / BM, 1), blk, 0, stream>>>(
      xB, D_DIM, 0, 0, W1B, F_DIM, 0, 0, b1, h1B, F_DIM, 0, 0,
      T_DIM, F_DIM, D_DIM, 1, 1.f);
  gemm_wmma<1, false, false, false><<<gDD, blk, 0, stream>>>(
      h1B, F_DIM, 0, 0, W2B, D_DIM, 0, 0, b2, ffF, D_DIM, 0, 0,
      T_DIM, D_DIM, F_DIM, 1, 1.f);

  // ---- out = LN(x + ffn) ----
  residual_ln_kernel<<<dim3(T_DIM), blk, 0, stream>>>(xF, ffF, g2, be2, outF, nullptr);

  (void)in_sizes; (void)n_in; (void)out_size; (void)ws_size;
}
